// ContactNet_4956392259996
// MI455X (gfx1250) — matrix operand order repaired and checked by reading, not compiled, not run on hardware
//
#include <hip/hip_runtime.h>

typedef __attribute__((ext_vector_type(16))) _Float16 v16h;
typedef __attribute__((ext_vector_type(8)))  _Float16 v8h;
typedef __attribute__((ext_vector_type(4)))  _Float16 v4h;
typedef __attribute__((ext_vector_type(8)))  float    v8f;

#define BB   64          // batch
#define NV   778         // verts
#define NP   784         // N padded to 16
#define NOBJ 2048

// ---------------------------------------------------------------- atomics
__device__ inline void atomicMaxF(float* addr, float val) {
    unsigned int* ua = (unsigned int*)addr;
    unsigned int old = *ua;
    while (true) {
        float f = __uint_as_float(old);
        if (f >= val) break;
        unsigned int assumed = old;
        old = atomicCAS(ua, assumed, __float_as_uint(val));
        if (old == assumed) break;
    }
}

// ------------------------------------------------- CDNA5 async global->LDS
__device__ inline void async_g2l_b128(unsigned lds_off, const void* gptr) {
    asm volatile("global_load_async_to_lds_b128 %0, %1, off"
                 :: "v"(lds_off), "v"((unsigned long long)(size_t)gptr)
                 : "memory");
}
__device__ inline void wait_asynccnt0() {
#if __has_builtin(__builtin_amdgcn_s_wait_asynccnt)
    __builtin_amdgcn_s_wait_asynccnt(0);
#else
    asm volatile("s_wait_asynccnt 0x0" ::: "memory");
#endif
}

// ---------------------------------------------------------------- WMMA GEMM
// C[b] = A[b] (MxK f16) * B[b] (KxN f16), N tiles of 16 along NP.
// 128 threads = 4 waves; each wave owns a 32x16 tile (2 accumulators).
// B panel (32x16) double-buffered in LDS, stored transposed [n][k] with
// pad=40 halves -> fragment read is two contiguous conflict-free ds_load_b128.
// Guards on M go through readfirstlane so v_wmma sits under a scalar branch
// (EXEC stays all-ones, per the WMMA ISA requirement).
// mode 0: d = relu(sc[row]*acc + bi[row]), zero cols >= nvalid, f16 store
// mode 1: bn then column-max via shfl_xor + atomicMaxF into gmax (pn3 pool)
// mode 2: d = relu(sc[col]*(acc + tadd[b*strideT+col]) + bi[col]), f16 store
__global__ __launch_bounds__(128) void gemm16_kernel(
    const _Float16* __restrict__ A, int lda, long long strideA,
    const _Float16* __restrict__ Bm, int ldb, long long strideB,
    int M, int K,
    _Float16* __restrict__ D, int ldc, long long strideD,
    const float* __restrict__ sc, const float* __restrict__ bi,
    const float* __restrict__ tadd, int strideT,
    float* __restrict__ gmax, int gstride,
    int mode, int nvalid)
{
    const int b    = blockIdx.z;
    const int n0   = blockIdx.x * 16;
    const int wv   = threadIdx.x >> 5;
    const int lane = threadIdx.x & 31;
    const int hh   = lane >> 4;     // which 16-lane half
    const int l    = lane & 15;
    const int m0   = blockIdx.y * 128 + wv * 32;

    const _Float16* Ab = A + (size_t)b * (size_t)strideA;
    const _Float16* Bb = Bm + (size_t)b * (size_t)strideB;

    __shared__ __align__(16) _Float16 Bs[2][16][40];

    v8f acc0 = {}; v8f acc1 = {};
    const int sAct0 = __builtin_amdgcn_readfirstlane((m0 < M) ? 1 : 0);
    const int sAct1 = __builtin_amdgcn_readfirstlane((m0 + 16 < M) ? 1 : 0);

    // cooperative B-panel stage: 512 halves, transposed [n][k] store
    auto stage = [&](int kk, int buf) {
        const int q = threadIdx.x * 4;
        const int k = q >> 4;
        const int n = q & 15;               // n, n+1, n+2, n+3 share k
        v4h vb = *(const v4h*)(Bb + (size_t)(kk + k) * ldb + (n0 + n));
        Bs[buf][n + 0][k] = vb[0];
        Bs[buf][n + 1][k] = vb[1];
        Bs[buf][n + 2][k] = vb[2];
        Bs[buf][n + 3][k] = vb[3];
    };

    stage(0, 0);
    __syncthreads();

    int pb = 0;
    for (int kk = 0; kk < K; kk += 32, pb ^= 1) {
        // B fragment: lane column = l, k = {hh*8..hh*8+7, 16+hh*8..16+hh*8+7}
        v8h b0 = *(const v8h*)&Bs[pb][l][hh * 8];
        v8h b1 = *(const v8h*)&Bs[pb][l][hh * 8 + 16];
        v16h bf = __builtin_shufflevector(b0, b1,
            0,1,2,3,4,5,6,7,8,9,10,11,12,13,14,15);

        if (kk + 32 < K) stage(kk + 32, pb ^ 1);   // overlap with WMMA

        if (sAct0) {  // A rows m0..m0+15: lane row = m0+l,
                      // k = {hh*8..hh*8+7, 16+hh*8..16+hh*8+7}
            const _Float16* ap = Ab + (size_t)(m0 + l) * lda + kk + hh * 8;
            v8h a0 = *(const v8h*)ap;
            v8h a1 = *(const v8h*)(ap + 16);
            v16h af = __builtin_shufflevector(a0, a1,
                0,1,2,3,4,5,6,7,8,9,10,11,12,13,14,15);
            acc0 = __builtin_amdgcn_wmma_f32_16x16x32_f16(
                false, af, false, bf, (short)0, acc0, false, false);
        }
        if (sAct1) {
            const _Float16* ap = Ab + (size_t)(m0 + 16 + l) * lda + kk + hh * 8;
            v8h a0 = *(const v8h*)ap;
            v8h a1 = *(const v8h*)(ap + 16);
            v16h af = __builtin_shufflevector(a0, a1,
                0,1,2,3,4,5,6,7,8,9,10,11,12,13,14,15);
            acc1 = __builtin_amdgcn_wmma_f32_16x16x32_f16(
                false, af, false, bf, (short)0, acc1, false, false);
        }
        __syncthreads();
    }

    // Epilogue. C/D layout: lane<16 -> col=lane, vgpr r = row r;
    //           lane>=16 -> col=lane-16, vgpr r = row 8+r.
    const int col = n0 + l;
    for (int f = 0; f < 2; ++f) {
        if (f == 0 ? !sAct0 : !sAct1) continue;
        v8f acc = (f == 0) ? acc0 : acc1;
        const int mb = m0 + f * 16;
        for (int r = 0; r < 8; ++r) {
            const int row = mb + hh * 8 + r;
            float v = acc[r];
            if (mode == 0) {
                v = sc[row] * v + bi[row];
                v = v > 0.f ? v : 0.f;
                if (col >= nvalid) v = 0.f;
                D[(size_t)b * strideD + (size_t)row * ldc + col] = (_Float16)v;
            } else if (mode == 1) {
                v = sc[row] * v + bi[row];
                if (col >= nvalid) v = -3.0e38f;
                v = fmaxf(v, __shfl_xor(v, 1, 32));
                v = fmaxf(v, __shfl_xor(v, 2, 32));
                v = fmaxf(v, __shfl_xor(v, 4, 32));
                v = fmaxf(v, __shfl_xor(v, 8, 32));
                if (l == 0) atomicMaxF(&gmax[(size_t)b * gstride + row], v);
            } else {
                float tv = tadd[(size_t)b * strideT + col];
                v = sc[col] * (v + tv) + bi[col];
                v = v > 0.f ? v : 0.f;
                D[(size_t)b * strideD + (size_t)row * ldc + col] = (_Float16)v;
            }
        }
    }
}

// ---------------------------------------------------------------- utilities
__global__ void convert_pad_kernel(const float* __restrict__ src, int rows, int cols,
                                   _Float16* __restrict__ dst, int dcols, int total)
{
    int i = blockIdx.x * blockDim.x + threadIdx.x;
    if (i >= total) return;
    int r = i / dcols, c = i % dcols;
    float v = (r < rows && c < cols) ? src[r * cols + c] : 0.f;
    dst[i] = (_Float16)v;
}

// WfT[c, o] = Wf[o, c] for c<778, o<778 (zero pad to 800x784)
__global__ void wft_kernel(const float* __restrict__ Wf, _Float16* __restrict__ dst)
{
    int i = blockIdx.x * blockDim.x + threadIdx.x;
    if (i >= 800 * NP) return;
    int c = i / NP, o = i % NP;
    float v = (c < NV && o < NV) ? Wf[o * 826 + c] : 0.f;
    dst[i] = (_Float16)v;
}

__global__ void rowsum_kernel(const float* __restrict__ Wf, float* __restrict__ rs)
{
    int o = blockIdx.x * blockDim.x + threadIdx.x;
    if (o >= NP) return;
    float s = 0.f;
    if (o < NV) for (int c = 0; c < NV; ++c) s += Wf[o * 826 + c];
    rs[o] = s;
}

// fold eval-mode BN: scale = g/sqrt(v+eps), bias = be + scale*(b-m)
__global__ void bnfold_kernel(const float* __restrict__ bp, const float* __restrict__ gp,
                              const float* __restrict__ bep, const float* __restrict__ mp,
                              const float* __restrict__ vp, int c, int cp,
                              float* __restrict__ sc, float* __restrict__ bi)
{
    int i = blockIdx.x * blockDim.x + threadIdx.x;
    if (i >= cp) return;
    if (i < c) {
        float s = gp[i] * rsqrtf(vp[i] + 1e-5f);
        sc[i] = s;
        bi[i] = bep[i] + s * (bp[i] - mp[i]);
    } else { sc[i] = 0.f; bi[i] = 0.f; }
}

__global__ void theta_kernel(const float* __restrict__ th, float* __restrict__ out)
{
    int i = blockIdx.x * blockDim.x + threadIdx.x;
    if (i >= BB * 48) return;
    out[i] = ((i % 48) < 3) ? 0.f : th[i];
}

// t[b,o] = sum_k Wf[o, 778+k] * theta_r[b,k]
__global__ void tvec_kernel(const float* __restrict__ Wf, const float* __restrict__ th,
                            float* __restrict__ t)
{
    int i = blockIdx.x * blockDim.x + threadIdx.x;
    if (i >= BB * NP) return;
    int b = i / NP, o = i % NP;
    float s = 0.f;
    if (o < NV) for (int k = 3; k < 48; ++k) s += Wf[o * 826 + NV + k] * th[b * 48 + k];
    t[i] = s;
}

// x_pc f16 [B, 32, NP]: rows 0..2 points^T, 3..5 normals^T, rest zero
__global__ void xpc_kernel(const float* __restrict__ P, const float* __restrict__ Nr,
                           _Float16* __restrict__ xpc)
{
    int i = blockIdx.x * blockDim.x + threadIdx.x;
    if (i >= BB * 32 * NP) return;
    int b = i / (32 * NP);
    int k = (i / NP) & 31;
    int n = i % NP;
    float v = 0.f;
    if (n < NV) {
        if (k < 3)      v = P[((size_t)b * NV + n) * 3 + k];
        else if (k < 6) v = Nr[((size_t)b * NV + n) * 3 + (k - 3)];
    }
    xpc[i] = (_Float16)v;
}

__global__ void ginit_kernel(float* __restrict__ g)
{
    int i = blockIdx.x * blockDim.x + threadIdx.x;
    if (i < BB * 1024) g[i] = -3.0e38f;
}

// U rows 0..1023: relu(sc_f[pt]*(rowsum[pt]*g[b,ch] + t[b,pt]) + bi_f[pt])
__global__ void upart_kernel(const float* __restrict__ g, const float* __restrict__ t,
                             const float* __restrict__ rs, const float* __restrict__ scf,
                             const float* __restrict__ bif, _Float16* __restrict__ U)
{
    long long i = (long long)blockIdx.x * blockDim.x + threadIdx.x;
    if (i >= (long long)BB * 1024 * NP) return;
    int b  = (int)(i / (1024 * NP));
    int ch = (int)((i / NP) % 1024);
    int pt = (int)(i % NP);
    float v = scf[pt] * (rs[pt] * g[b * 1024 + ch] + t[b * NP + pt]) + bif[pt];
    v = v > 0.f ? v : 0.f;
    U[(size_t)b * 1088 * NP + (size_t)ch * NP + pt] = (_Float16)v;
}

// c4 (128 -> 1, no BN) + sigmoid
__global__ void c4sig_kernel(const _Float16* __restrict__ V3, const float* __restrict__ W4,
                             const float* __restrict__ b4, float* __restrict__ out0)
{
    int i = blockIdx.x * blockDim.x + threadIdx.x;
    if (i >= BB * NP) return;
    int b = i / NP, pt = i % NP;
    if (pt >= NV) return;
    float acc = b4[0];
    for (int c = 0; c < 128; ++c)
        acc += W4[c] * (float)V3[(size_t)b * 128 * NP + (size_t)c * NP + pt];
    out0[b * NV + pt] = 1.f / (1.f + __expf(-acc));
}

// ---------------------------------------------------------------- contacts
__global__ __launch_bounds__(256) void contact_kernel(
    const float* __restrict__ probs, const float* __restrict__ verts,
    float* __restrict__ cpad, int* __restrict__ counts,
    float* __restrict__ center, float* __restrict__ mind)
{
    const int b = blockIdx.x;
    const int t = threadIdx.x;
    __shared__ unsigned char msk[NP];
    __shared__ int scanb[256];
    __shared__ int total;
    __shared__ float csum[3];
    __shared__ float red[256];

    int cl = 0;
    for (int j = 0; j < 4; ++j) {
        int i = 4 * t + j;
        int m = 0;
        if (i < NV) m = (probs[b * NV + i] >= 0.5f) ? 1 : 0;
        if (i < NP) msk[i] = (unsigned char)m;
        cl += m;
    }
    scanb[t] = cl;
    __syncthreads();
    for (int off = 1; off < 256; off <<= 1) {
        int add = (t >= off) ? scanb[t - off] : 0;
        __syncthreads();
        scanb[t] += add;
        __syncthreads();
    }
    if (t == 255) total = scanb[255];
    __syncthreads();
    if (total == 0) {   // fallback: last 2 verts
        if (t == 0) { msk[776] = 1; msk[777] = 1; }
        __syncthreads();
        cl = 0;
        for (int j = 0; j < 4; ++j) { int i = 4 * t + j; if (i < NV) cl += msk[i]; }
        scanb[t] = cl;
        __syncthreads();
        for (int off = 1; off < 256; off <<= 1) {
            int add = (t >= off) ? scanb[t - off] : 0;
            __syncthreads();
            scanb[t] += add;
            __syncthreads();
        }
        if (t == 255) total = scanb[255];
        __syncthreads();
    }
    if (t == 0) { csum[0] = 0.f; csum[1] = 0.f; csum[2] = 0.f; }
    __syncthreads();
    int pos = scanb[t] - cl;           // stable exclusive prefix
    float sx = 0.f, sy = 0.f, sz = 0.f;
    for (int j = 0; j < 4; ++j) {
        int i = 4 * t + j;
        if (i < NV && msk[i]) {
            float x = verts[((size_t)b * NV + i) * 3 + 0];
            float y = verts[((size_t)b * NV + i) * 3 + 1];
            float z = verts[((size_t)b * NV + i) * 3 + 2];
            cpad[((size_t)b * NV + pos) * 3 + 0] = x;
            cpad[((size_t)b * NV + pos) * 3 + 1] = y;
            cpad[((size_t)b * NV + pos) * 3 + 2] = z;
            sx += x; sy += y; sz += z;
            ++pos;
        }
    }
    for (int r = total + t; r < NV; r += 256) {
        cpad[((size_t)b * NV + r) * 3 + 0] = 0.f;
        cpad[((size_t)b * NV + r) * 3 + 1] = 0.f;
        cpad[((size_t)b * NV + r) * 3 + 2] = 0.f;
    }
    atomicAdd(&csum[0], sx);
    atomicAdd(&csum[1], sy);
    atomicAdd(&csum[2], sz);
    __syncthreads();
    float cnt = (float)total;
    float cx = csum[0] / cnt, cy = csum[1] / cnt, cz = csum[2] / cnt;
    float md = 3.0e38f;
    for (int j = 0; j < 4; ++j) {
        int i = 4 * t + j;
        if (i < NV && msk[i]) {
            float dx = verts[((size_t)b * NV + i) * 3 + 0] - cx;
            float dy = verts[((size_t)b * NV + i) * 3 + 1] - cy;
            float dz = verts[((size_t)b * NV + i) * 3 + 2] - cz;
            md = fminf(md, sqrtf(dx * dx + dy * dy + dz * dz));
        }
    }
    red[t] = md;
    __syncthreads();
    for (int o = 128; o > 0; o >>= 1) {
        if (t < o) red[t] = fminf(red[t], red[t + o]);
        __syncthreads();
    }
    if (t == 0) {
        counts[b] = total;
        center[b * 3 + 0] = cx; center[b * 3 + 1] = cy; center[b * 3 + 2] = cz;
        mind[b] = red[0];
    }
}

__global__ void maxcount_kernel(const int* __restrict__ counts, int* __restrict__ maxc)
{
    __shared__ int s[64];
    int t = threadIdx.x;
    s[t] = counts[t];
    __syncthreads();
    for (int o = 32; o > 0; o >>= 1) {
        if (t < o) s[t] = max(s[t], s[t + o]);
        __syncthreads();
    }
    if (t == 0) maxc[0] = s[0];
}

__device__ inline unsigned int pcg_step(unsigned int& st)
{
    st = st * 747796405u + 2891336453u;
    unsigned int w = ((st >> ((st >> 28) + 4u)) ^ st) * 277803737u;
    return (w >> 22) ^ w;
}

// sphere-surface-ish random points (deterministic hash stand-in for jax PRNG)
__global__ void objgen_kernel(const float* __restrict__ center, const float* __restrict__ mind,
                              const int* __restrict__ counts, const int* __restrict__ maxc,
                              float* __restrict__ out2)
{
    int i = blockIdx.x * blockDim.x + threadIdx.x;
    if (i >= BB * NOBJ) return;
    int b = i / NOBJ, q = i % NOBJ;
    unsigned int st = 42u * 2654435761u ^ (unsigned int)(i * 0x9E3779B9u);
    float v0 = (pcg_step(st) >> 8) * (1.f / 16777216.f) - 0.5f;
    float v1 = (pcg_step(st) >> 8) * (1.f / 16777216.f) - 0.5f;
    float v2 = (pcg_step(st) >> 8) * (1.f / 16777216.f) - 0.5f;
    float nrm = sqrtf(v0 * v0 + v1 * v1 + v2 * v2);
    float s = mind[b] / (nrm + 1e-6f);
    out2[(size_t)i * 4 + 0] = v0 * s + center[b * 3 + 0];
    out2[(size_t)i * 4 + 1] = v1 * s + center[b * 3 + 1];
    out2[(size_t)i * 4 + 2] = v2 * s + center[b * 3 + 2];
    out2[(size_t)i * 4 + 3] = (q == 0 && counts[b] < maxc[0]) ? 1.f : 0.f;
}

// 1-NN of each valid contact vert against 2048 obj points, scatter feat.
// One workgroup per batch: the 32KB object table (2048 x 16B rows) is staged
// into LDS with CDNA5 async global->LDS B128 copies (ASYNCcnt-tracked), then
// every lane scans it with broadcast LDS reads.
__global__ __launch_bounds__(256) void knn_kernel(
    const float* __restrict__ cpad, const int* __restrict__ counts,
    float* __restrict__ out2)
{
    const int b = blockIdx.x;
    const int t = threadIdx.x;
    __shared__ __align__(16) float objs[NOBJ * 4];    // 32 KB
    const unsigned lbase = (unsigned)(size_t)&objs[0];
    for (int j = 0; j < NOBJ / 256; ++j) {
        const int row = t + 256 * j;
        async_g2l_b128(lbase + (unsigned)row * 16u,
                       (const void*)(out2 + ((size_t)b * NOBJ + row) * 4));
    }
    wait_asynccnt0();
    __syncthreads();

    const int cnt = counts[b];
    for (int p = t; p < NV; p += 256) {
        if (p >= cnt) continue;
        float px = cpad[((size_t)b * NV + p) * 3 + 0];
        float py = cpad[((size_t)b * NV + p) * 3 + 1];
        float pz = cpad[((size_t)b * NV + p) * 3 + 2];
        float best = 3.0e38f;
        int bq = 0;
        for (int q = 0; q < NOBJ; ++q) {
            float dx = px - objs[q * 4 + 0];
            float dy = py - objs[q * 4 + 1];
            float dz = pz - objs[q * 4 + 2];
            float d = dx * dx + dy * dy + dz * dz;
            if (d < best) { best = d; bq = q; }
        }
        best = fmaxf(best, 0.f);
        float val = 2.f / (1.f + __expf(best));       // 2*sigmoid(-d)
        out2[((size_t)b * NOBJ + bq) * 4 + 3] = val;
    }
}

// ---------------------------------------------------------------- host
extern "C" void kernel_launch(void* const* d_in, const int* in_sizes, int n_in,
                              void* d_out, int out_size, void* d_ws, size_t ws_size,
                              hipStream_t stream)
{
    (void)in_sizes; (void)n_in; (void)out_size; (void)ws_size;
    // inputs (params pytree flattened alphabetically: W,b,be,g,m,v per layer;
    // layers c1,c2,c3,c4,fuse,pn1,pn2,pn3)
    const float* theta   = (const float*)d_in[0];
    const float* vertsN  = (const float*)d_in[1];
    const float* vertsRP = (const float*)d_in[2];
    const float* vertsRN = (const float*)d_in[3];
    const float* c1W = (const float*)d_in[4];
    const float* c1b = (const float*)d_in[5];
    const float* c1be= (const float*)d_in[6];
    const float* c1g = (const float*)d_in[7];
    const float* c1m = (const float*)d_in[8];
    const float* c1v = (const float*)d_in[9];
    const float* c2W = (const float*)d_in[10];
    const float* c2b = (const float*)d_in[11];
    const float* c2be= (const float*)d_in[12];
    const float* c2g = (const float*)d_in[13];
    const float* c2m = (const float*)d_in[14];
    const float* c2v = (const float*)d_in[15];
    const float* c3W = (const float*)d_in[16];
    const float* c3b = (const float*)d_in[17];
    const float* c3be= (const float*)d_in[18];
    const float* c3g = (const float*)d_in[19];
    const float* c3m = (const float*)d_in[20];
    const float* c3v = (const float*)d_in[21];
    const float* c4W = (const float*)d_in[22];
    const float* c4b = (const float*)d_in[23];
    const float* fW  = (const float*)d_in[24];
    const float* fb  = (const float*)d_in[25];
    const float* fbe = (const float*)d_in[26];
    const float* fg  = (const float*)d_in[27];
    const float* fm  = (const float*)d_in[28];
    const float* fv  = (const float*)d_in[29];
    const float* p1W = (const float*)d_in[30];
    const float* p1b = (const float*)d_in[31];
    const float* p1be= (const float*)d_in[32];
    const float* p1g = (const float*)d_in[33];
    const float* p1m = (const float*)d_in[34];
    const float* p1v = (const float*)d_in[35];
    const float* p2W = (const float*)d_in[36];
    const float* p2b = (const float*)d_in[37];
    const float* p2be= (const float*)d_in[38];
    const float* p2g = (const float*)d_in[39];
    const float* p2m = (const float*)d_in[40];
    const float* p2v = (const float*)d_in[41];
    const float* p3W = (const float*)d_in[42];
    const float* p3b = (const float*)d_in[43];
    const float* p3be= (const float*)d_in[44];
    const float* p3g = (const float*)d_in[45];
    const float* p3m = (const float*)d_in[46];
    const float* p3v = (const float*)d_in[47];

    float* out0 = (float*)d_out;                       // probs [64,778]
    float* out1 = out0 + (size_t)BB * NV;              // contact_pad [64,778,3]
    float* out2 = out1 + (size_t)BB * NV * 3;          // object_base [64,2048,4]
    float* out3 = out2 + (size_t)BB * NOBJ * 4;        // theta_r [64,48]

    // ---- carve workspace
    char* wsb = (char*)d_ws;
    size_t off = 0;
    auto carve = [&](size_t bytes) -> void* {
        void* p = (void*)(wsb + off);
        off += (bytes + 255) & ~(size_t)255;
        return p;
    };
    _Float16* w1h  = (_Float16*)carve((size_t)64 * 32 * 2);
    _Float16* w2h  = (_Float16*)carve((size_t)128 * 64 * 2);
    _Float16* w3h  = (_Float16*)carve((size_t)1024 * 128 * 2);
    _Float16* wfT  = (_Float16*)carve((size_t)800 * NP * 2);
    _Float16* wc1h = (_Float16*)carve((size_t)512 * 1088 * 2);
    _Float16* wc2h = (_Float16*)carve((size_t)256 * 512 * 2);
    _Float16* wc3h = (_Float16*)carve((size_t)128 * 256 * 2);
    _Float16* xpc  = (_Float16*)carve((size_t)BB * 32 * NP * 2);
    _Float16* h1   = (_Float16*)carve((size_t)BB * 64 * 800 * 2);
    _Float16* h2   = (_Float16*)carve((size_t)BB * 128 * NP * 2);
    _Float16* U    = (_Float16*)carve((size_t)BB * 1088 * NP * 2);
    _Float16* V1   = (_Float16*)carve((size_t)BB * 512 * NP * 2);
    _Float16* V2   = U;                                        // alias: U dead after c1
    _Float16* V3   = (_Float16*)((char*)U + (size_t)26 * 1024 * 1024); // alias (disjoint from V2)
    float* g    = (float*)carve((size_t)BB * 1024 * 4);
    float* tmat = (float*)carve((size_t)BB * NP * 4);
    float* rs   = (float*)carve((size_t)NP * 4);
    float* s1 = (float*)carve(64*4);    float* bi1 = (float*)carve(64*4);
    float* s2 = (float*)carve(128*4);   float* bi2 = (float*)carve(128*4);
    float* s3 = (float*)carve(1024*4);  float* bi3 = (float*)carve(1024*4);
    float* sf = (float*)carve(NP*4);    float* bif = (float*)carve(NP*4);
    float* sc1= (float*)carve(512*4);   float* bc1 = (float*)carve(512*4);
    float* sc2= (float*)carve(256*4);   float* bc2 = (float*)carve(256*4);
    float* sc3= (float*)carve(128*4);   float* bc3 = (float*)carve(128*4);
    int*   counts = (int*)carve(BB*4);
    float* centerb= (float*)carve(BB*3*4);
    float* mindb  = (float*)carve(BB*4);
    int*   maxc   = (int*)carve(4);

    auto UT = [&](long long total) { return dim3((unsigned)((total + 255) / 256)); };

    // ---- weight prep (f16 conversions, padding, BN folds)
    convert_pad_kernel<<<UT(64*32), 256, 0, stream>>>(p1W, 64, 6, w1h, 32, 64*32);
    convert_pad_kernel<<<UT(128*64), 256, 0, stream>>>(p2W, 128, 64, w2h, 64, 128*64);
    convert_pad_kernel<<<UT(1024*128), 256, 0, stream>>>(p3W, 1024, 128, w3h, 128, 1024*128);
    convert_pad_kernel<<<UT(512*1088), 256, 0, stream>>>(c1W, 512, 1088, wc1h, 1088, 512*1088);
    convert_pad_kernel<<<UT(256*512), 256, 0, stream>>>(c2W, 256, 512, wc2h, 512, 256*512);
    convert_pad_kernel<<<UT(128*256), 256, 0, stream>>>(c3W, 128, 256, wc3h, 256, 128*256);
    wft_kernel<<<UT(800*NP), 256, 0, stream>>>(fW, wfT);
    rowsum_kernel<<<UT(NP), 256, 0, stream>>>(fW, rs);
    bnfold_kernel<<<UT(64), 256, 0, stream>>>(p1b, p1g, p1be, p1m, p1v, 64, 64, s1, bi1);
    bnfold_kernel<<<UT(128), 256, 0, stream>>>(p2b, p2g, p2be, p2m, p2v, 128, 128, s2, bi2);
    bnfold_kernel<<<UT(1024), 256, 0, stream>>>(p3b, p3g, p3be, p3m, p3v, 1024, 1024, s3, bi3);
    bnfold_kernel<<<UT(NP), 256, 0, stream>>>(fb, fg, fbe, fm, fv, NV, NP, sf, bif);
    bnfold_kernel<<<UT(512), 256, 0, stream>>>(c1b, c1g, c1be, c1m, c1v, 512, 512, sc1, bc1);
    bnfold_kernel<<<UT(256), 256, 0, stream>>>(c2b, c2g, c2be, c2m, c2v, 256, 256, sc2, bc2);
    bnfold_kernel<<<UT(128), 256, 0, stream>>>(c3b, c3g, c3be, c3m, c3v, 128, 128, sc3, bc3);
    theta_kernel<<<UT(BB*48), 256, 0, stream>>>(theta, out3);
    tvec_kernel<<<UT((long long)BB*NP), 256, 0, stream>>>(fW, theta, tmat);
    xpc_kernel<<<UT((long long)BB*32*NP), 256, 0, stream>>>(vertsRP, vertsRN, xpc);
    ginit_kernel<<<UT(BB*1024), 256, 0, stream>>>(g);
    hipMemsetAsync(h1, 0, (size_t)BB * 64 * 800 * 2, stream);   // zero K-pad cols 784..799

    const dim3 blk(128);
    // pn1: h1 = relu(bn(W1 @ xpc))   [64 x 784], ld 800
    gemm16_kernel<<<dim3(NP/16, 1, BB), blk, 0, stream>>>(
        w1h, 32, 0, xpc, NP, (long long)32*NP, 64, 32,
        h1, 800, (long long)64*800, s1, bi1, nullptr, 0, nullptr, 0, 0, NV);
    // pn2: h2 = relu(bn(W2 @ h1))    [128 x 784]
    gemm16_kernel<<<dim3(NP/16, 1, BB), blk, 0, stream>>>(
        w2h, 64, 0, h1, 800, (long long)64*800, 128, 64,
        h2, NP, (long long)128*NP, s2, bi2, nullptr, 0, nullptr, 0, 0, NV);
    // pn3 + fused max-pool: g[b,ch] = max_n bn(W3 @ h2)
    gemm16_kernel<<<dim3(NP/16, 8, BB), blk, 0, stream>>>(
        w3h, 128, 0, h2, NP, (long long)128*NP, 1024, 128,
        U, NP, 0, s3, bi3, nullptr, 0, g, 1024, 1, NV);
    // U rows 0..1023 from the rank-1 (rowsum(Wf) x g) + theta-term structure
    upart_kernel<<<UT((long long)BB*1024*NP), 256, 0, stream>>>(g, tmat, rs, sf, bif, U);
    // fuse real part: U rows 1024..1087 = relu(bn_col(h1 @ WfT + t))
    gemm16_kernel<<<dim3(NP/16, 1, BB), blk, 0, stream>>>(
        h1, 800, (long long)64*800, wfT, NP, 0, 64, 800,
        U + (size_t)1024*NP, NP, (long long)1088*NP, sf, bif, tmat, NP, nullptr, 0, 2, NV);
    // c1: V1 = relu(bn(Wc1 @ U))   K=1088
    gemm16_kernel<<<dim3(NP/16, 4, BB), blk, 0, stream>>>(
        wc1h, 1088, 0, U, NP, (long long)1088*NP, 512, 1088,
        V1, NP, (long long)512*NP, sc1, bc1, nullptr, 0, nullptr, 0, 0, NV);
    // c2: V2 = relu(bn(Wc2 @ V1))  (V2 aliases dead U)
    gemm16_kernel<<<dim3(NP/16, 2, BB), blk, 0, stream>>>(
        wc2h, 512, 0, V1, NP, (long long)512*NP, 256, 512,
        V2, NP, (long long)256*NP, sc2, bc2, nullptr, 0, nullptr, 0, 0, NV);
    // c3: V3 = relu(bn(Wc3 @ V2))
    gemm16_kernel<<<dim3(NP/16, 1, BB), blk, 0, stream>>>(
        wc3h, 256, 0, V2, NP, (long long)256*NP, 128, 256,
        V3, NP, (long long)128*NP, sc3, bc3, nullptr, 0, nullptr, 0, 0, NV);
    // c4 + sigmoid -> probs (out0)
    c4sig_kernel<<<UT((long long)BB*NP), 256, 0, stream>>>(V3, c4W, c4b, out0);
    // contact extraction -> out1, counts/center/min_dist
    contact_kernel<<<dim3(BB), dim3(256), 0, stream>>>(out0, vertsN, out1, counts, centerb, mindb);
    maxcount_kernel<<<dim3(1), dim3(64), 0, stream>>>(counts, maxc);
    // object points + feat init -> out2
    objgen_kernel<<<UT((long long)BB*NOBJ), 256, 0, stream>>>(centerb, mindb, counts, maxc, out2);
    // 1-NN with LDS-staged object table (async global->LDS)
    knn_kernel<<<dim3(BB), dim3(256), 0, stream>>>(out1, counts, out2);
}
